// KernelUpdateHead_37555194036866
// MI455X (gfx1250) — matrix operand order repaired and checked by reading, not compiled, not run on hardware
//
#include <hip/hip_runtime.h>
#include <hip/hip_bf16.h>
#include <math.h>

// Problem constants (reference: B,N,C,H,W = 4,64,256,128,128)
#define BQ   4
#define NQ   64
#define CQ   256
#define HQ   128
#define WQ   128
#define HWQ  (HQ*WQ)
#define NHQ  8
#define HDQ  32
#define FFNHQ 2048
#define EPSQ 1e-5f

typedef __attribute__((ext_vector_type(16))) __bf16 v16bf;
typedef __attribute__((ext_vector_type(8)))  __bf16 v8bf;
typedef __attribute__((ext_vector_type(8)))  float v8f;

static __device__ __forceinline__ unsigned pack_bf16(float a, float b)
{
    union { __bf16 h[2]; unsigned u; } x;
    x.h[0] = (__bf16)a; x.h[1] = (__bf16)b;   // -> v_cvt_pk_bf16_f32
    return x.u;
}

// ---------------------------------------------------------------------------
// Generic batched WMMA GEMM:  D[b] = act(A[b] * B[b] + bias),  D is f32.
//   A: [M x K] row-major (lda, batch stride sA).  M % 32 == 0, K % 64 == 0.
//   TRANS=1: Bsrc is [Nn x K] row-major (x @ W.T);  TRANS=0: Bsrc is [K x Nn].
//   GUARDN: branchless clamp+select for Nn not a multiple of 64.
//   ABF / BBF: A / B global arrays are already bf16 (else f32, packed on the fly).
//   KSPLIT>1: gridDim.y = (M/32)*KSPLIT; each slice accumulates its K-chunk
//             into pre-zeroed D via f32 global atomics (bias/act must be off).
// Block = 128 threads = 4 waves; block tile = 32(M) x 64(N) x 64(K) per sync;
// each wave owns a 32x16 slab -> 4 v_wmma per barrier pair.
// ---------------------------------------------------------------------------
template<bool TRANS, bool GUARDN, bool ABF, bool BBF, int KSPLIT>
__global__ void k_wmma_gemm(const void* __restrict__ Av,
                            const void* __restrict__ Bv,
                            const float* __restrict__ bias,
                            float* __restrict__ D,
                            int M, int Nn, int K,
                            int lda, int ldb, int ldd,
                            long long sA, long long sB, long long sD,
                            int act)
{
    // A: m-major [32][64] (128B rows). B: n-major [64][72] (144B rows, both
    // 16B-aligned so a lane's fragment chunks are ds_load_b128s).
    __shared__ alignas(16) __bf16 As[32][64];
    __shared__ alignas(16) __bf16 Bs[64][72];
    unsigned* AsU = (unsigned*)&As[0][0];   // row stride 32 dwords
    unsigned* BsU = (unsigned*)&Bs[0][0];   // row stride 36 dwords

    const float*  Af = (const float*)Av;
    const __bf16* Ah = (const __bf16*)Av;
    const float*  Bf = (const float*)Bv;
    const __bf16* Bh = (const __bf16*)Bv;

    const int tid  = threadIdx.x;
    const int lane = tid & 31;
    const int wave = tid >> 5;

    const long long bz = blockIdx.z;
    const long long aoff = bz * sA, boff = bz * sB;
    D += bz * sD;

    int my, ks;
    if (KSPLIT > 1) {
        const int mtiles = M >> 5;
        my = blockIdx.y % mtiles;
        ks = blockIdx.y / mtiles;
    } else { my = blockIdx.y; ks = 0; }
    const int m0 = my * 32;
    const int n0 = blockIdx.x * 64;
    const int kchunk = K / KSPLIT;
    const int kbeg = ks * kchunk, kend = kbeg + kchunk;

    v8f acc0 = {0.f,0.f,0.f,0.f,0.f,0.f,0.f,0.f};
    v8f acc1 = {0.f,0.f,0.f,0.f,0.f,0.f,0.f,0.f};

    for (int k0 = kbeg; k0 < kend; k0 += 64) {
        // ---- stage A [32 m x 64 k]: 1024 bf16-pairs, 8 dwords per thread
        #pragma unroll
        for (int i = 0; i < 8; ++i) {
            int pidx = tid + 128*i;            // 0..1023
            int m  = pidx >> 5;                // 0..31
            int kp = pidx & 31;                // pair index (k = 2*kp)
            long long off = aoff + (long long)(m0 + m)*lda + k0 + 2*kp;
            unsigned pk;
            if (ABF) pk = *(const unsigned*)(Ah + off);
            else     { const float* ap = Af + off; pk = pack_bf16(ap[0], ap[1]); }
            AsU[m*32 + kp] = pk;
        }
        // ---- stage B [64 n x 64 k] into n-major LDS
        if (TRANS) {
            #pragma unroll
            for (int i = 0; i < 16; ++i) {
                int pidx = tid + 128*i;        // 0..2047
                int n  = pidx >> 5;            // 0..63
                int kp = pidx & 31;
                int gn = n0 + n;
                int gnc = GUARDN ? min(gn, Nn - 1) : gn;
                long long off = boff + (long long)gnc*ldb + k0 + 2*kp;
                unsigned pk;
                if (BBF) pk = *(const unsigned*)(Bh + off);
                else     { float2 v = *(const float2*)(Bf + off); pk = pack_bf16(v.x, v.y); }
                if (GUARDN && gn >= Nn) pk = 0u;           // v_cndmask
                BsU[n*36 + kp] = pk;
            }
        } else {
            #pragma unroll
            for (int i = 0; i < 16; ++i) {
                int pidx = tid + 128*i;        // 0..2047
                int kk = pidx >> 5;            // 0..63
                int np = pidx & 31;            // n pair -> n = 2np, 2np+1
                int gn = n0 + 2*np;
                int gnc = GUARDN ? min(gn, Nn - 2) : gn;
                long long off = boff + (long long)(k0 + kk)*ldb + gnc;
                __bf16 h0, h1;
                if (BBF) {
                    union { unsigned u; __bf16 h[2]; } u;
                    u.u = *(const unsigned*)(Bh + off);
                    h0 = u.h[0]; h1 = u.h[1];
                } else {
                    float2 v = *(const float2*)(Bf + off);
                    h0 = (__bf16)v.x; h1 = (__bf16)v.y;
                }
                if (GUARDN && gn >= Nn) { h0 = (__bf16)0.f; h1 = (__bf16)0.f; }
                Bs[2*np    ][kk] = h0;
                Bs[2*np + 1][kk] = h1;
            }
        }
        // speculative near-cache prefetch of next A K-tile (element-correct)
        {
            long long poff = aoff + (long long)(m0 + (tid & 31))*lda + (k0 + 64);
            const void* pp = ABF ? (const void*)(Ah + poff) : (const void*)(Af + poff);
            __builtin_prefetch(pp, 0, 3);
        }
        __syncthreads();

        // ---- 2 K-substeps x 2 M-tiles = 4 WMMA per barrier pair
        #pragma unroll
        for (int kss = 0; kss < 2; ++kss) {
            const int kb  = 32*kss;
            const int mf  = lane & 15;
            const int akb = kb + (lane >> 4) * 16;  // lanes0-15: K0-15; 16-31: K16-31
            v8bf a00 = *(const v8bf*)&As[mf     ][akb];
            v8bf a01 = *(const v8bf*)&As[mf     ][akb + 8];
            v8bf a10 = *(const v8bf*)&As[16 + mf][akb];
            v8bf a11 = *(const v8bf*)&As[16 + mf][akb + 8];
            v16bf afa = __builtin_shufflevector(a00, a01, 0,1,2,3,4,5,6,7,8,9,10,11,12,13,14,15);
            v16bf afb = __builtin_shufflevector(a10, a11, 0,1,2,3,4,5,6,7,8,9,10,11,12,13,14,15);
            const int nf  = wave*16 + (lane & 15);
            const int bkb = kb + (lane >> 4) * 16;  // lanes0-15: K0-15; 16-31: K16-31
            v16bf bf_ = *(const v16bf*)&Bs[nf][bkb];  // 32B -> 2x ds_load_b128

            acc0 = __builtin_amdgcn_wmma_f32_16x16x32_bf16(
                       false, afa, false, bf_, (short)0, acc0, false, false);
            acc1 = __builtin_amdgcn_wmma_f32_16x16x32_bf16(
                       false, afb, false, bf_, (short)0, acc1, false, false);
        }
        __syncthreads();
    }

    // ---- store: D VGPR r holds M = r + 8*(lane>=16), N = lane&15
    const int nc = n0 + wave*16 + (lane & 15);
    if (!GUARDN || nc < Nn) {
        const int mrb = m0 + ((lane >> 4) * 8);
        if (KSPLIT > 1) {
            // accumulate K-slice into pre-zeroed D: global_atomic_add_f32
            #pragma unroll
            for (int r = 0; r < 8; ++r) {
                __hip_atomic_fetch_add(&D[(long long)(mrb + r     )*ldd + nc], acc0[r],
                                       __ATOMIC_RELAXED, __HIP_MEMORY_SCOPE_AGENT);
                __hip_atomic_fetch_add(&D[(long long)(mrb + r + 16)*ldd + nc], acc1[r],
                                       __ATOMIC_RELAXED, __HIP_MEMORY_SCOPE_AGENT);
            }
        } else {
            float bv = bias ? bias[nc] : 0.f;
            #pragma unroll
            for (int r = 0; r < 8; ++r) {
                float v0 = acc0[r] + bv;
                float v1 = acc1[r] + bv;
                if (act == 1) { v0 = fmaxf(v0, 0.f); v1 = fmaxf(v1, 0.f); }
                D[(long long)(mrb + r     )*ldd + nc] = v0;
                D[(long long)(mrb + r + 16)*ldd + nc] = v1;
            }
        }
    }
}

// ---------------------------------------------------------------------------
__global__ void k_zero(float* __restrict__ p, long long n)
{
    for (long long i = (long long)blockIdx.x*256 + threadIdx.x; i < n;
         i += (long long)gridDim.x*256) p[i] = 0.f;
}

// GroupNorm stats: one block per (b, group); 8 channels x HW per group.
__global__ void k_gn_stats(const float* __restrict__ xf,
                           float* __restrict__ mean, float* __restrict__ inv)
{
    __shared__ float s1[256], s2[256];
    const int bg = blockIdx.x;           // b*32 + g
    const int b = bg >> 5, g = bg & 31;
    const float* p = xf + ((long long)b*CQ + g*8) * HWQ;
    float s = 0.f, ss = 0.f;
    for (int i = threadIdx.x; i < 8*HWQ; i += 256) {
        float v = p[i]; s += v; ss += v*v;
    }
    s1[threadIdx.x] = s; s2[threadIdx.x] = ss; __syncthreads();
    for (int st = 128; st > 0; st >>= 1) {
        if (threadIdx.x < st) { s1[threadIdx.x] += s1[threadIdx.x+st];
                                s2[threadIdx.x] += s2[threadIdx.x+st]; }
        __syncthreads();
    }
    if (threadIdx.x == 0) {
        const float cnt = 8.f * HWQ;
        float m = s1[0] / cnt;
        float var = s2[0] / cnt - m*m;
        mean[bg] = m; inv[bg] = rsqrtf(var + EPSQ);
    }
}

// GN apply + ReLU, f32 in -> packed bf16 out (pairs stay inside one channel).
__global__ void k_gn_apply(const float* __restrict__ xfp, __bf16* __restrict__ xfh,
                           const float* __restrict__ mean, const float* __restrict__ inv,
                           const float* __restrict__ gg, const float* __restrict__ gb)
{
    const long long np = (long long)BQ*CQ*HWQ/2;
    for (long long i = (long long)blockIdx.x*256 + threadIdx.x; i < np;
         i += (long long)gridDim.x*256) {
        long long e = 2*i;
        int b  = (int)(e / ((long long)CQ*HWQ));
        int c  = (int)((e / HWQ) % CQ);
        int bg = b*32 + (c >> 3);
        float2 v = *(const float2*)(xfp + e);
        float m = mean[bg], iv = inv[bg], g = gg[c], bb = gb[c];
        float r0 = fmaxf((v.x - m)*iv*g + bb, 0.f);
        float r1 = fmaxf((v.y - m)*iv*g + bb, 0.f);
        *(unsigned*)(xfh + e) = pack_bf16(r0, r1);
    }
}

// gm = (mask_preds > 0) as bf16 {0,1}, packed pairs.
__global__ void k_gm(const float* __restrict__ mp, __bf16* __restrict__ gm)
{
    const long long np = (long long)BQ*NQ*HWQ/2;
    for (long long i = (long long)blockIdx.x*256 + threadIdx.x; i < np;
         i += (long long)gridDim.x*256) {
        long long e = 2*i;
        float2 v = *(const float2*)(mp + e);
        *(unsigned*)(gm + e) = pack_bf16(v.x > 0.f ? 1.f : 0.f,
                                         v.y > 0.f ? 1.f : 0.f);
    }
}

// LayerNorm over C=256; optional residual in2; optional trailing ReLU.
__global__ void k_ln(const float* __restrict__ in1, const float* __restrict__ in2,
                     const float* __restrict__ g, const float* __restrict__ b,
                     float* __restrict__ out, int relu)
{
    __shared__ float s1[256], s2[256];
    const int row = blockIdx.x, c = threadIdx.x;
    float v = in1[(long long)row*256 + c];
    if (in2) v += in2[(long long)row*256 + c];
    s1[c] = v; s2[c] = v*v; __syncthreads();
    for (int st = 128; st > 0; st >>= 1) {
        if (c < st) { s1[c] += s1[c+st]; s2[c] += s2[c+st]; }
        __syncthreads();
    }
    float m = s1[0] * (1.f/256.f);
    float var = s2[0] * (1.f/256.f) - m*m;
    float o = (v - m) * rsqrtf(var + EPSQ) * g[c] + b[c];
    if (relu) o = fmaxf(o, 0.f);
    out[(long long)row*256 + c] = o;
}

// dysep: d[b,m,c] = relu(dw0*v[c-1] + dw1*v[c] + dw2*v[c+1]) (zero-padded)
__global__ void k_dysep_d(const float* __restrict__ w, const float* __restrict__ val,
                          float* __restrict__ d)
{
    const int i = blockIdx.x*256 + threadIdx.x;   // 0..65535
    const int c = i & 255;
    const int row = i >> 8;                        // b*64 + m
    const float* wr = w + (long long)row*67;       // [dw0 dw1 dw2 | pw...]
    const float* vr = val + (long long)row*256;
    float v = wr[1]*vr[c];
    if (c > 0)   v += wr[0]*vr[c-1];
    if (c < 255) v += wr[2]*vr[c+1];
    d[i] = fmaxf(v, 0.f);
}

// MHA core with online softmax; one block per (b, head), one thread per query.
__global__ void k_attn(const float* __restrict__ qkv, float* __restrict__ out)
{
    __shared__ float Ks[NQ][HDQ];
    __shared__ float Vs[NQ][HDQ];
    const int b = blockIdx.x / NHQ, h = blockIdx.x % NHQ;
    const int t = threadIdx.x;                     // 0..63 (query index)
    const float* base = qkv + (long long)(b*NQ) * (3*CQ);
    for (int d = 0; d < HDQ; ++d) {
        Ks[t][d] = base[(long long)t*(3*CQ) +   CQ + h*HDQ + d];
        Vs[t][d] = base[(long long)t*(3*CQ) + 2*CQ + h*HDQ + d];
    }
    __syncthreads();
    float q[HDQ];
    const float scale = 0.17677669529663687f;      // 32^-0.5
    for (int d = 0; d < HDQ; ++d)
        q[d] = base[(long long)t*(3*CQ) + h*HDQ + d] * scale;
    float mx = -1e30f, sm = 0.f, o[HDQ];
    for (int d = 0; d < HDQ; ++d) o[d] = 0.f;
    for (int n = 0; n < NQ; ++n) {
        float s = 0.f;
        for (int d = 0; d < HDQ; ++d) s += q[d] * Ks[n][d];
        float nm = fmaxf(mx, s);
        float corr = __expf(mx - nm);
        float e = __expf(s - nm);
        sm = sm*corr + e;
        for (int d = 0; d < HDQ; ++d) o[d] = o[d]*corr + e*Vs[n][d];
        mx = nm;
    }
    float rs = 1.f / sm;
    float* op = out + (long long)(b*NQ + t)*CQ + h*HDQ;
    for (int d = 0; d < HDQ; ++d) op[d] = o[d]*rs;
}

// Bilinear 2x upsample (half-pixel centers, edge clamp), 128->256.
__global__ void k_upsample(const float* __restrict__ in, float* __restrict__ out)
{
    const long long total = (long long)BQ*NQ*(2*HQ)*(2*WQ);
    for (long long i = (long long)blockIdx.x*256 + threadIdx.x; i < total;
         i += (long long)gridDim.x*256) {
        int x = (int)(i % (2*WQ));
        long long r = i / (2*WQ);
        int y = (int)(r % (2*HQ));
        long long bn = r / (2*HQ);
        float sy = (y + 0.5f)*0.5f - 0.5f;
        float sx = (x + 0.5f)*0.5f - 0.5f;
        int y0 = (int)floorf(sy); float fy = sy - (float)y0;
        int x0 = (int)floorf(sx); float fx = sx - (float)x0;
        int y1 = y0 + 1, x1 = x0 + 1;
        y0 = max(0, min(HQ-1, y0)); y1 = max(0, min(HQ-1, y1));
        x0 = max(0, min(WQ-1, x0)); x1 = max(0, min(WQ-1, x1));
        const float* p = in + bn*(long long)(HQ*WQ);
        float v00 = p[y0*WQ + x0], v01 = p[y0*WQ + x1];
        float v10 = p[y1*WQ + x0], v11 = p[y1*WQ + x1];
        out[i] = v00*(1.f-fy)*(1.f-fx) + v01*(1.f-fy)*fx
               + v10*fy*(1.f-fx)       + v11*fy*fx;
    }
}

// ---------------------------------------------------------------------------
static inline void gemm_f32(hipStream_t s,
                            const float* A, const float* B, const float* bias, float* D,
                            int M, int Nn, int K, int lda, int ldb, int ldd,
                            long long sA, long long sB, long long sD,
                            int batches, int bTrans, int act)
{
    dim3 g((Nn + 63)/64, M/32, batches);
    const bool guard = (Nn % 64) != 0;
    if (bTrans) {
        if (guard) k_wmma_gemm<true, true ,false,false,1><<<g,128,0,s>>>(A,B,bias,D,M,Nn,K,lda,ldb,ldd,sA,sB,sD,act);
        else       k_wmma_gemm<true, false,false,false,1><<<g,128,0,s>>>(A,B,bias,D,M,Nn,K,lda,ldb,ldd,sA,sB,sD,act);
    } else {
        if (guard) k_wmma_gemm<false,true ,false,false,1><<<g,128,0,s>>>(A,B,bias,D,M,Nn,K,lda,ldb,ldd,sA,sB,sD,act);
        else       k_wmma_gemm<false,false,false,false,1><<<g,128,0,s>>>(A,B,bias,D,M,Nn,K,lda,ldb,ldd,sA,sB,sD,act);
    }
}

extern "C" void kernel_launch(void* const* d_in, const int* in_sizes, int n_in,
                              void* d_out, int out_size, void* d_ws, size_t ws_size,
                              hipStream_t stream)
{
    // ---- inputs (setup_inputs dict order, params flattened depth-first) ----
    const float* x  = (const float*)d_in[0];
    const float* pf = (const float*)d_in[1];
    const float* mp = (const float*)d_in[2];
    int p = 3;
#define NEXT ((const float*)d_in[p++])
    const float* loc_w = NEXT;
    const float* gn_g = NEXT;  const float* gn_b = NEXT;
    const float* f_wl_w = NEXT; const float* f_wl_b = NEXT;
    const float* f_ln_g = NEXT; const float* f_ln_b = NEXT;
    const float* k_wl_w = NEXT; const float* k_wl_b = NEXT;
    const float* k_ln_g = NEXT; const float* k_ln_b = NEXT;
    const float* f_norm_g = NEXT; const float* f_norm_b = NEXT;
    const float* k_norm_g = NEXT; const float* k_norm_b = NEXT;
    const float* in_w = NEXT;  const float* in_b = NEXT;
    const float* out_w = NEXT; const float* out_b = NEXT;
    const float* s_norm_g = NEXT; const float* s_norm_b = NEXT;
    const float* fc1_w = NEXT; const float* fc1_b = NEXT;
    const float* fc2_w = NEXT; const float* fc2_b = NEXT;
    const float* ffn_ln_g = NEXT; const float* ffn_ln_b = NEXT;
    const float* mfc_w[3] = {NEXT, NEXT, NEXT};
    const float* mln_g[3] = {NEXT, NEXT, NEXT};
    const float* mln_b[3] = {NEXT, NEXT, NEXT};
    const float* fcm_w = NEXT; const float* fcm_b = NEXT;
#undef NEXT
    (void)in_sizes; (void)n_in; (void)out_size; (void)ws_size;

    // ---- workspace carve-out (float units) ----
    float* ws = (float*)d_ws;
    long long o = 0;
    float*  xfpre = ws + o; o += (long long)BQ*CQ*HWQ;       // 16,777,216 f32
    float*  preds = xfpre;                                   // reuse after GN
    __bf16* xfh   = (__bf16*)(ws + o); o += (long long)BQ*CQ*HWQ/2;   // bf16 xf
    __bf16* gmh   = (__bf16*)(ws + o); o += (long long)BQ*NQ*HWQ/2;   // bf16 mask
    float* stm   = ws + o; o += 128;
    float* sti   = ws + o; o += 128;
    float* topk  = ws + o; o += 65536;
    float* w1    = ws + o; o += 17408;                     // [256 x 67]
    float* d1    = ws + o; o += 65536;
    float* dys1  = ws + o; o += 65536;
    float* dys1n = ws + o; o += 65536;
    float* t1    = ws + o; o += 65536;
    float* w2    = ws + o; o += 17408;
    float* d2    = ws + o; o += 65536;
    float* dys2  = ws + o; o += 65536;
    float* dys2n = ws + o; o += 65536;
    float* t2    = ws + o; o += 65536;
    float* qkv   = ws + o; o += (long long)BQ*NQ*3*CQ;     // 196,608
    float* atto  = ws + o; o += 65536;
    float* tmp   = ws + o; o += 65536;
    float* t3    = ws + o; o += 65536;
    float* ffnh  = ws + o; o += (long long)BQ*NQ*FFNHQ;    // 524,288
    float* tmp2  = ws + o; o += 65536;
    float* t4    = ws + o; o += 65536;
    float* mA    = ws + o; o += 65536;
    float* mB    = ws + o; o += 65536;

    float* out_preds = (float*)d_out;
    float* out_mf    = out_preds + (long long)BQ*NQ*4*HQ*WQ;  // +16,777,216

    const long long sCHW = (long long)CQ*HWQ;
    const long long sNHW = (long long)NQ*HWQ;
    const long long sNC  = (long long)NQ*CQ;
    const int rows = BQ*NQ;   // 256 flat token rows

    // 1) xf_pre = loc_w @ x   (per-batch [256x256]x[256x16384], f32 x f32)
    gemm_f32(stream, loc_w, x, nullptr, xfpre, CQ, HWQ, CQ, CQ, HWQ, HWQ,
             0, sCHW, sCHW, BQ, /*bTrans=*/0, /*act=*/0);
    // 2-3) GroupNorm(32) + ReLU -> bf16 xf
    k_gn_stats<<<BQ*32, 256, 0, stream>>>(xfpre, stm, sti);
    k_gn_apply<<<8192, 256, 0, stream>>>(xfpre, xfh, stm, sti, gn_g, gn_b);
    // 4) binary mask -> bf16
    k_gm<<<4096, 256, 0, stream>>>(mp, gmh);
    // 5) topk[b] = gm[b] @ xf[b]^T  (M=64, K=16384, N=256; bf16 x bf16,
    //    split-K x8 -> 256 workgroups, accumulate via global_atomic_add_f32)
    k_zero<<<64, 256, 0, stream>>>(topk, (long long)BQ*NQ*CQ);
    {
        dim3 g(CQ/64, (NQ/32)*8, BQ);
        k_wmma_gemm<true,false,true,true,8><<<g,128,0,stream>>>(
            gmh, xfh, nullptr, topk, NQ, CQ, HWQ, HWQ, HWQ, CQ,
            sNHW, sCHW, sNC, 0);
    }

    // 6) dysep #1 (f_atten), value = topk
    gemm_f32(stream, pf, f_wl_w, f_wl_b, w1, rows, NQ+3, CQ, CQ, CQ, NQ+3,
             0, 0, 0, 1, 1, 0);
    k_dysep_d<<<256, 256, 0, stream>>>(w1, topk, d1);
    gemm_f32(stream, w1 + 3, d1, nullptr, dys1, NQ, CQ, NQ, NQ+3, CQ, CQ,
             (long long)NQ*(NQ+3), sNC, sNC, BQ, 0, 0);
    k_ln<<<rows, 256, 0, stream>>>(dys1, nullptr, f_ln_g, f_ln_b, dys1n, 0);
    k_ln<<<rows, 256, 0, stream>>>(topk, dys1n, f_norm_g, f_norm_b, t1, 0);

    // 7) dysep #2 (k_atten), value = t1
    gemm_f32(stream, pf, k_wl_w, k_wl_b, w2, rows, NQ+3, CQ, CQ, CQ, NQ+3,
             0, 0, 0, 1, 1, 0);
    k_dysep_d<<<256, 256, 0, stream>>>(w2, t1, d2);
    gemm_f32(stream, w2 + 3, d2, nullptr, dys2, NQ, CQ, NQ, NQ+3, CQ, CQ,
             (long long)NQ*(NQ+3), sNC, sNC, BQ, 0, 0);
    k_ln<<<rows, 256, 0, stream>>>(dys2, nullptr, k_ln_g, k_ln_b, dys2n, 0);
    k_ln<<<rows, 256, 0, stream>>>(t1, dys2n, k_norm_g, k_norm_b, t2, 0);

    // 8) MHA
    gemm_f32(stream, t2, in_w, in_b, qkv, rows, 3*CQ, CQ, CQ, CQ, 3*CQ,
             0, 0, 0, 1, 1, 0);
    k_attn<<<BQ*NHQ, NQ, 0, stream>>>(qkv, atto);
    gemm_f32(stream, atto, out_w, out_b, tmp, rows, CQ, CQ, CQ, CQ, CQ,
             0, 0, 0, 1, 1, 0);
    k_ln<<<rows, 256, 0, stream>>>(tmp, t2, s_norm_g, s_norm_b, t3, 0);

    // 9) FFN (pre) + LN
    gemm_f32(stream, t3, fc1_w, fc1_b, ffnh, rows, FFNHQ, CQ, CQ, CQ, FFNHQ,
             0, 0, 0, 1, 1, /*relu=*/1);
    gemm_f32(stream, ffnh, fc2_w, fc2_b, tmp2, rows, CQ, FFNHQ, FFNHQ, FFNHQ, CQ,
             0, 0, 0, 1, 1, 0);
    k_ln<<<rows, 256, 0, stream>>>(tmp2, t3, ffn_ln_g, ffn_ln_b, t4, 0);

    // 10) mask FC chain: 3x (GEMM -> LN -> ReLU)
    const float* cur = t4;
    for (int i = 0; i < 3; ++i) {
        gemm_f32(stream, cur, mfc_w[i], nullptr, mA, rows, CQ, CQ, CQ, CQ, CQ,
                 0, 0, 0, 1, 1, 0);
        k_ln<<<rows, 256, 0, stream>>>(mA, nullptr, mln_g[i], mln_b[i], mB, /*relu=*/1);
        cur = mB;
    }
    // 11) mask_feat -> second output region (also A of the preds GEMM)
    gemm_f32(stream, cur, fcm_w, fcm_b, out_mf, rows, CQ, CQ, CQ, CQ, CQ,
             0, 0, 0, 1, 1, 0);
    // 12) preds[b] = mask_feat[b] @ xf[b]   (M=64, K=256, N=16384; f32 x bf16)
    {
        dim3 g(HWQ/64, NQ/32, BQ);
        k_wmma_gemm<false,false,false,true,1><<<g,128,0,stream>>>(
            out_mf, xfh, nullptr, preds, NQ, HWQ, CQ, CQ, HWQ, HWQ,
            sNC, sCHW, sNHW, 0);
    }
    // 13) bilinear 2x upsample -> first output region
    k_upsample<<<16384, 256, 0, stream>>>(preds, out_preds);
}
